// GraphSAGE_18906446037603
// MI455X (gfx1250) — matrix operand order repaired and checked by reading, not compiled, not run on hardware
//
#include <hip/hip_runtime.h>
#include <hip/hip_bf16.h>
#include <math.h>

// GraphSAGE-mean x2 + L2 normalize for MI455X (gfx1250, wave32, WMMA).
// Pipeline: zero -> pack weights (bf16 hi/lo, B-fragment order) -> degree ->
// scatter(feat) -> WMMA gemm1 (self + mean, bias, relu) -> zero -> scatter(h1)
// -> WMMA gemm2 -> row normalize.
// GEMMs use bf16x3 split (Ah*Bh + Ah*Bl + Al*Bh, fp32 accum) ~ fp32 accuracy.
// Each wave processes TPW col tiles so A load+split cost is amortized 4x.

typedef __bf16 bf16_t;
typedef __attribute__((ext_vector_type(16))) __bf16 v16bf;
typedef __attribute__((ext_vector_type(8)))  float  v8f;

#define DIN 256
#define KDIM 256          // K is 256 for both layers

__device__ __forceinline__ void split_bf16(float x, __bf16& h, __bf16& l) {
    float fhi = __uint_as_float(__float_as_uint(x) & 0xFFFF0000u);
    h = (__bf16)fhi;          // exact (low mantissa bits zero)
    l = (__bf16)(x - fhi);    // residual, exact subtraction
}

// ---------------- utility kernels ----------------

__global__ void zero_kernel(float4* __restrict__ p, long long n4) {
    long long i = (long long)blockIdx.x * blockDim.x + threadIdx.x;
    if (i < n4) p[i] = make_float4(0.f, 0.f, 0.f, 0.f);
}

__global__ void degree_kernel(const int* __restrict__ dst, float* __restrict__ deg, int E) {
    int e = blockIdx.x * blockDim.x + threadIdx.x;
    if (e < E) atomicAdd(deg + dst[e], 1.0f);
}

// scatter-add: agg[dst] += X[src], 4 floats per thread, lanes cover one row chunk
__global__ void scatter_kernel(const float* __restrict__ X,
                               const int* __restrict__ src,
                               const int* __restrict__ dst,
                               float* __restrict__ agg,
                               int E, int D) {
    long long tid = (long long)blockIdx.x * blockDim.x + threadIdx.x;
    int chunks = D >> 2;
    if (tid >= (long long)E * chunks) return;
    int e = (int)(tid / chunks);
    int c = (int)(tid % chunks) << 2;
    int s = src[e], d = dst[e];
    float4 v = *(const float4*)(X + (size_t)s * D + c);
    float* o = agg + (size_t)d * D + c;
    atomicAdd(o + 0, v.x);
    atomicAdd(o + 1, v.y);
    atomicAdd(o + 2, v.z);
    atomicAdd(o + 3, v.w);
}

// Pack weight matrix W[K x Nout] (row-major) into WMMA B-fragment order,
// split into bf16 hi/lo. Fragment index = ((ks*ntiles+ct)*32 + lane)*16 + elem,
// with k = ks*32 + (lane>=16 ? 8 : 0) + (elem&7) + (elem>=8 ? 16 : 0), n = ct*16 + (lane&15).
__global__ void pack_weights_kernel(const float* __restrict__ w, int K, int Nout,
                                    bf16_t* __restrict__ hi, bf16_t* __restrict__ lo) {
    int tid = blockIdx.x * blockDim.x + threadIdx.x;
    if (tid >= K * Nout) return;
    int ntiles = Nout >> 4;
    int elem =  tid        & 15;
    int lane = (tid >> 4)  & 31;
    int frag =  tid >> 9;
    int ct = frag % ntiles;
    int ks = frag / ntiles;
    int k = ks * 32 + ((lane >> 4) << 3) + (elem & 7) + ((elem >> 3) << 4);
    int n = ct * 16 + (lane & 15);
    __bf16 h, l;
    split_bf16(w[(size_t)k * Nout + n], h, l);
    hi[tid] = h;
    lo[tid] = l;
}

// ---------------- WMMA GEMM ----------------

// Load 16 bf16x2 A-fragment elements for one 16x32 K-step from row-major fp32,
// scaled by `scale`, split hi/lo.  Per ISA 16-bit A layout:
//   lane row = lane&15 ; kbase = (lane>>4)*8
//   elems 0..7  -> K = kbase + {0..7}
//   elems 8..15 -> K = kbase + 16 + {0..7}
__device__ __forceinline__ void load_split_A(const float* __restrict__ p, float scale,
                                             v16bf& h, v16bf& l) {
    float4 x0 = *(const float4*)(p);
    float4 x1 = *(const float4*)(p + 4);
    float4 x2 = *(const float4*)(p + 16);
    float4 x3 = *(const float4*)(p + 20);
    float t[16] = {x0.x, x0.y, x0.z, x0.w, x1.x, x1.y, x1.z, x1.w,
                   x2.x, x2.y, x2.z, x2.w, x3.x, x3.y, x3.z, x3.w};
#pragma unroll
    for (int i = 0; i < 16; ++i) {
        float v = t[i] * scale;
        __bf16 hh, ll;
        split_bf16(v, hh, ll);
        h[i] = hh;
        l[i] = ll;
    }
}

// Out[row, n] = relu( Aself @ Wself + (Agg/deg) @ Wneigh + bias )
// Block = 128 (4 waves); wave w owns col tiles [w*TPW, w*TPW+TPW).
// Grid.x = row tiles (M/16).  4 waves * TPW must equal Nout/16.
template <int TPW>
__global__ __launch_bounds__(128)
void sage_gemm_kernel(const float* __restrict__ Aself,
                      const float* __restrict__ Agg,
                      const float* __restrict__ deg,
                      const bf16_t* __restrict__ BsH, const bf16_t* __restrict__ BsL,
                      const bf16_t* __restrict__ BnH, const bf16_t* __restrict__ BnL,
                      const float* __restrict__ bias,
                      float* __restrict__ Out,
                      int Nout) {
    const int lane = threadIdx.x & 31;
    const int wave = threadIdx.x >> 5;
    const int rt = blockIdx.x;
    const int ct0 = wave * TPW;
    const int ntiles = Nout >> 4;

    const int arow = rt * 16 + (lane & 15);
    const int kg = (lane >> 4) * 8;
    const float inv = __builtin_amdgcn_rcpf(fmaxf(deg[arow], 1.0f));

    v8f acc[TPW];
#pragma unroll
    for (int t = 0; t < TPW; ++t) acc[t] = (v8f){};

#pragma unroll
    for (int ks = 0; ks < KDIM / 32; ++ks) {
        const float* pa = Aself + (size_t)arow * KDIM + ks * 32 + kg;
        const float* pm = Agg   + (size_t)arow * KDIM + ks * 32 + kg;
        v16bf ash, asl, anh, anl;
        load_split_A(pa, 1.0f, ash, asl);
        load_split_A(pm, inv,  anh, anl);

#pragma unroll
        for (int t = 0; t < TPW; ++t) {
            const int ct = ct0 + t;
            size_t foff = ((size_t)(ks * ntiles + ct) * 32 + lane) * 16;
            v16bf bsh = *(const v16bf*)(BsH + foff);
            v16bf bsl = *(const v16bf*)(BsL + foff);
            v16bf bnh = *(const v16bf*)(BnH + foff);
            v16bf bnl = *(const v16bf*)(BnL + foff);

            acc[t] = __builtin_amdgcn_wmma_f32_16x16x32_bf16(false, ash, false, bsh, (short)0, acc[t], false, false);
            acc[t] = __builtin_amdgcn_wmma_f32_16x16x32_bf16(false, ash, false, bsl, (short)0, acc[t], false, false);
            acc[t] = __builtin_amdgcn_wmma_f32_16x16x32_bf16(false, asl, false, bsh, (short)0, acc[t], false, false);
            acc[t] = __builtin_amdgcn_wmma_f32_16x16x32_bf16(false, anh, false, bnh, (short)0, acc[t], false, false);
            acc[t] = __builtin_amdgcn_wmma_f32_16x16x32_bf16(false, anh, false, bnl, (short)0, acc[t], false, false);
            acc[t] = __builtin_amdgcn_wmma_f32_16x16x32_bf16(false, anl, false, bnh, (short)0, acc[t], false, false);
        }
    }

    // D layout: vgpr v, lanes 0-15 -> M=v, N=lane ; lanes 16-31 -> M=v+8, N=lane-16
    const int mbase = rt * 16 + ((lane >> 4) << 3);
#pragma unroll
    for (int t = 0; t < TPW; ++t) {
        const int n = (ct0 + t) * 16 + (lane & 15);
        const float bn = bias[n];
#pragma unroll
        for (int v = 0; v < 8; ++v) {
            float x = acc[t][v] + bn;
            x = fmaxf(x, 0.0f);                   // relu (both layers)
            Out[(size_t)(mbase + v) * Nout + n] = x;
        }
    }
}

// L2-normalize rows of Out [Nrows x 128] in place. One wave per row.
__global__ __launch_bounds__(256)
void normalize_kernel(float* __restrict__ out, int Nrows) {
    const int lane = threadIdx.x & 31;
    const int wave = threadIdx.x >> 5;
    const int row = blockIdx.x * 8 + wave;
    if (row >= Nrows) return;
    float* p = out + (size_t)row * 128;
    float4 v = *(const float4*)(p + lane * 4);
    float ss = v.x * v.x + v.y * v.y + v.z * v.z + v.w * v.w;
#pragma unroll
    for (int off = 16; off > 0; off >>= 1) ss += __shfl_xor(ss, off, 32);
    float scale = 1.0f / fmaxf(sqrtf(ss), 1e-12f);
    v.x *= scale; v.y *= scale; v.z *= scale; v.w *= scale;
    *(float4*)(p + lane * 4) = v;
}

// ---------------- host launch ----------------

extern "C" void kernel_launch(void* const* d_in, const int* in_sizes, int n_in,
                              void* d_out, int out_size, void* d_ws, size_t ws_size,
                              hipStream_t stream) {
    const float* feat     = (const float*)d_in[0];
    const int*   src      = (const int*)  d_in[1];
    const int*   dst      = (const int*)  d_in[2];
    const float* w_self1  = (const float*)d_in[3];
    const float* w_neigh1 = (const float*)d_in[4];
    const float* b1       = (const float*)d_in[5];
    const float* w_self2  = (const float*)d_in[6];
    const float* w_neigh2 = (const float*)d_in[7];
    const float* b2       = (const float*)d_in[8];

    const int N    = in_sizes[0] / DIN;   // 100000
    const int E    = in_sizes[1];         // 800000
    const int DH   = in_sizes[5];         // 256
    const int DOUT = in_sizes[8];         // 128

    // workspace layout (fp32 elements)
    float* ws  = (float*)d_ws;
    float* deg = ws;                              // N
    float* agg = ws + N;                          // N*256
    float* h1  = agg + (size_t)N * DIN;           // N*256
    bf16_t* pk = (bf16_t*)(h1 + (size_t)N * DH);  // packed weights
    bf16_t* ws1h = pk;
    bf16_t* ws1l = ws1h + (size_t)DIN * DH;
    bf16_t* wn1h = ws1l + (size_t)DIN * DH;
    bf16_t* wn1l = wn1h + (size_t)DIN * DH;
    bf16_t* ws2h = wn1l + (size_t)DIN * DH;
    bf16_t* ws2l = ws2h + (size_t)DH * DOUT;
    bf16_t* wn2h = ws2l + (size_t)DH * DOUT;
    bf16_t* wn2l = wn2h + (size_t)DH * DOUT;

    float* out = (float*)d_out;

    const int B = 256;

    // 1. zero deg + agg (contiguous)
    {
        long long n4 = ((long long)N * (1 + DIN)) >> 2;
        int blocks = (int)((n4 + B - 1) / B);
        zero_kernel<<<blocks, B, 0, stream>>>((float4*)deg, n4);
    }
    // 2. pack weights into B-fragment order (hi/lo bf16)
    {
        int n1 = DIN * DH, n2 = DH * DOUT;
        pack_weights_kernel<<<(n1 + B - 1) / B, B, 0, stream>>>(w_self1,  DIN, DH,   ws1h, ws1l);
        pack_weights_kernel<<<(n1 + B - 1) / B, B, 0, stream>>>(w_neigh1, DIN, DH,   wn1h, wn1l);
        pack_weights_kernel<<<(n2 + B - 1) / B, B, 0, stream>>>(w_self2,  DH,  DOUT, ws2h, ws2l);
        pack_weights_kernel<<<(n2 + B - 1) / B, B, 0, stream>>>(w_neigh2, DH,  DOUT, wn2h, wn2l);
    }
    // 3. in-degree
    degree_kernel<<<(E + B - 1) / B, B, 0, stream>>>(dst, deg, E);
    // 4. scatter layer-1 messages: agg[dst] += feat[src]
    {
        long long t = (long long)E * (DIN >> 2);
        int blocks = (int)((t + B - 1) / B);
        scatter_kernel<<<blocks, B, 0, stream>>>(feat, src, dst, agg, E, DIN);
    }
    // 5. layer-1 GEMM: h1 = relu(feat@Ws1 + mean@Wn1 + b1)   [100000 x 256]
    //    4 waves x TPW=4 col tiles = 16 tiles = 256 cols
    sage_gemm_kernel<4><<<dim3(N / 16), 128, 0, stream>>>(feat, agg, deg,
                                                          ws1h, ws1l, wn1h, wn1l,
                                                          b1, h1, DH);
    // 6. re-zero agg for layer 2
    {
        long long n4 = ((long long)N * DH) >> 2;
        int blocks = (int)((n4 + B - 1) / B);
        zero_kernel<<<blocks, B, 0, stream>>>((float4*)agg, n4);
    }
    // 7. scatter layer-2 messages: agg[dst] += h1[src]
    {
        long long t = (long long)E * (DH >> 2);
        int blocks = (int)((t + B - 1) / B);
        scatter_kernel<<<blocks, B, 0, stream>>>(h1, src, dst, agg, E, DH);
    }
    // 8. layer-2 GEMM: out = relu(h1@Ws2 + mean@Wn2 + b2)    [100000 x 128]
    //    4 waves x TPW=2 col tiles = 8 tiles = 128 cols
    sage_gemm_kernel<2><<<dim3(N / 16), 128, 0, stream>>>(h1, agg, deg,
                                                          ws2h, ws2l, wn2h, wn2l,
                                                          b2, out, DOUT);
    // 9. row L2-normalize in place
    normalize_kernel<<<(N + 7) / 8, B, 0, stream>>>(out, N);
}
